// GraphTextReranker_30562987278952
// MI455X (gfx1250) — matrix operand order repaired and checked by reading, not compiled, not run on hardware
//
#include <hip/hip_runtime.h>
#include <hip/hip_bf16.h>
#include <cstdint>
#include <cstddef>

// ---------------- problem constants (from reference) ----------------
#define DG     256        // graph feature dim
#define NATOM  2048
#define NEDGE  8192
#define BATCH  32
#define TTXT   128
#define HDIM   768
#define NHEAD  12
#define DFF    3072
#define NLAYER 12
#define TOPK   64
#define SEQ    195        // 1 + 64 + 1 + 128 + 1
#define NTOK   (BATCH * SEQ)   // 6240

// ---------------- CDNA5 WMMA types ----------------
typedef __attribute__((ext_vector_type(16))) __bf16 bf16x16;
typedef __attribute__((ext_vector_type(8)))  float  f32x8;
typedef __attribute__((ext_vector_type(4)))  unsigned int u32x4;
typedef __attribute__((ext_vector_type(8)))  int i32x8;
typedef __attribute__((ext_vector_type(4)))  int i32x4;

union FragU { bf16x16 v; unsigned int u[8]; };

__device__ __forceinline__ unsigned int pack2bf(float a, float b) {
  unsigned int ua = __builtin_bit_cast(unsigned int, a);
  unsigned int ub = __builtin_bit_cast(unsigned int, b);
  ua = (ua + 0x7FFFu + ((ua >> 16) & 1u)) >> 16;   // RNE f32 -> bf16
  ub = (ub + 0x7FFFu + ((ub >> 16) & 1u)) >> 16;
  return (ua & 0xFFFFu) | (ub << 16);
}
__device__ __forceinline__ float bf2f(unsigned int lo16) {
  unsigned int u = lo16 << 16;
  return __builtin_bit_cast(float, u);
}
__device__ __forceinline__ float gelu_f(float x) {
  return 0.5f * x * (1.0f + erff(x * 0.70710678118654752f));
}
__device__ __forceinline__ float apply_act(float x, int act) {
  if (act == 1) return gelu_f(x);
  if (act == 2) return fmaxf(x, 0.0f);
  return x;
}

// ---- TDM: DMA one (16 x 64-uint) tile of packed-bf16 weights into LDS ----
// D# group0/1 per CDNA5 ISA ch.8: data_size=4B, 2D tile 64x16,
// LDS padding: 1 dword after every 64 dwords -> [16][65] padded rows.
__device__ __forceinline__ void tdm_load_b_tile(unsigned ldsOff, const unsigned* gsrc,
                                                unsigned dim0_rem, unsigned dim1_rem,
                                                unsigned strideU)
{
  unsigned long long ga = (unsigned long long)(size_t)gsrc;
  u32x4 g0;
  g0.x = 1u;                                             // count = 1 valid descriptor
  g0.y = ldsOff;                                         // LDS byte address
  g0.z = (unsigned)(ga & 0xFFFFFFFFu);                   // global_addr[31:0]
  g0.w = (unsigned)((ga >> 32) & 0x1FFFFFFu) | (2u << 30); // global_addr[56:32], type=2
  i32x8 g1;
  g1[0] = (int)((2u << 16) | (1u << 20) | (5u << 22));   // data_size=4B, pad_en, every 64dw, +1dw
  g1[1] = (int)((dim0_rem & 0xFFFFu) << 16);             // tensor_dim0[15:0]
  g1[2] = (int)((dim0_rem >> 16) | ((dim1_rem & 0xFFFFu) << 16)); // dim0[31:16], dim1[15:0]
  g1[3] = (int)((dim1_rem >> 16) | (64u << 16));         // dim1[31:16], tile_dim0=64
  g1[4] = 16;                                            // tile_dim1=16, tile_dim2=0
  g1[5] = (int)strideU;                                  // tensor_dim0_stride[31:0]
  g1[6] = 0;
  g1[7] = 0;
  i32x4 z4 = {};
#if defined(__clang_major__) && (__clang_major__ >= 23)
  i32x8 z8 = {};
  __builtin_amdgcn_tensor_load_to_lds(g0, g1, z4, z4, z8, 0);
#else
  __builtin_amdgcn_tensor_load_to_lds(g0, g1, z4, z4, 0);
#endif
}

// pack f32 weight (K,N) row-major -> uint (K/2, N): {bf16(W[2kp][n]), bf16(W[2kp+1][n])}
__global__ __launch_bounds__(256) void packw_k(const float* __restrict__ W,
                                               unsigned int* __restrict__ Wp,
                                               int K, int N) {
  long i = (long)blockIdx.x * 256 + threadIdx.x;
  long total = (long)(K / 2) * N;
  if (i >= total) return;
  int kp = (int)(i / N), n = (int)(i % N);
  Wp[i] = pack2bf(W[(size_t)(2 * kp) * N + n], W[(size_t)(2 * kp + 1) * N + n]);
}

// ---------------- bf16-WMMA GEMM: C = act(A[M,K] @ W[K,N] + bias) ----------
// Tile 128x64, BK=32, 256 threads = 8 waves in 4x2; each wave -> 32x32 (4 wmma accs)
// B tiles arrive via Tensor Data Mover from pre-packed bf16 weights.
#define BM 128
#define BN 64
#define BK 32

__global__ __launch_bounds__(256) void gemm_wmma_k(
    const float* __restrict__ A, const unsigned int* __restrict__ Wp,
    const float* __restrict__ bias, float* __restrict__ C,
    int M, int N, int K, int act)
{
  __shared__ unsigned int As[BM][BK / 2 + 1];   // K packed in bf16 pairs
  __shared__ unsigned int Bs[BK / 2][BN + 1];   // TDM-filled (pad 1 dword / row)

  const int tid  = threadIdx.x;
  const int lane = tid & 31;
  const int wave = tid >> 5;            // 0..7
  const int wm   = wave >> 1;           // 0..3  (row group of 32)
  const int wn   = wave & 1;            // 0..1  (col group of 32)
  const int half = lane >> 4;
  const int l16  = lane & 15;

  const int m0 = blockIdx.y * BM;
  const int n0 = blockIdx.x * BN;
  const int Kp = K / 2;
  const unsigned ldsB = (unsigned)(size_t)&Bs[0][0];

  f32x8 acc00 = {}; f32x8 acc01 = {}; f32x8 acc10 = {}; f32x8 acc11 = {};

  const int numK = K / BK;
  for (int kt = 0; kt < numK; ++kt) {
    const int k0 = kt * BK;
    // kick off the B-panel DMA (wave0 only; TENSORcnt tracks it)
    if (wave == 0)
      tdm_load_b_tile(ldsB, Wp + (size_t)(kt * 16) * N + n0,
                      (unsigned)(N - n0), (unsigned)(Kp - kt * 16), (unsigned)N);
    // stage A tile (128 x 32 f32 -> bf16 pairs), overlapping the DMA
    for (int i = tid; i < BM * 8; i += 256) {
      int row = i >> 3, c4 = i & 7;
      float4 f;
      if (m0 + row < M) {
        f = *reinterpret_cast<const float4*>(A + (size_t)(m0 + row) * K + k0 + c4 * 4);
        if (kt + 1 < numK)
          __builtin_prefetch((const void*)(A + (size_t)(m0 + row) * K + k0 + BK + c4 * 4), 0, 1);
      } else {
        f = make_float4(0.f, 0.f, 0.f, 0.f);
      }
      As[row][c4 * 2]     = pack2bf(f.x, f.y);
      As[row][c4 * 2 + 1] = pack2bf(f.z, f.w);
    }
    if (wave == 0) __builtin_amdgcn_s_wait_tensorcnt(0);
    __syncthreads();

    FragU a0, a1, b0, b1;
#pragma unroll
    for (int v = 0; v < 8; ++v) {
      // A 16x32 bf16 layout: v<4 -> kpair v + 4*half ; v>=4 -> kpair 4+v+4*half
      int kpA = (v < 4) ? (v + 4 * half) : (4 + v + 4 * half);
      a0.u[v] = As[wm * 32 + l16][kpA];
      a1.u[v] = As[wm * 32 + 16 + l16][kpA];
      // B 32x16 bf16 layout: same kpair as A, col = l16
      int kpB = kpA;
      b0.u[v] = Bs[kpB][wn * 32 + l16];
      b1.u[v] = Bs[kpB][wn * 32 + 16 + l16];
    }
    acc00 = __builtin_amdgcn_wmma_f32_16x16x32_bf16(false, a0.v, false, b0.v, (short)0, acc00, false, false);
    acc01 = __builtin_amdgcn_wmma_f32_16x16x32_bf16(false, a0.v, false, b1.v, (short)0, acc01, false, false);
    acc10 = __builtin_amdgcn_wmma_f32_16x16x32_bf16(false, a1.v, false, b0.v, (short)0, acc10, false, false);
    acc11 = __builtin_amdgcn_wmma_f32_16x16x32_bf16(false, a1.v, false, b1.v, (short)0, acc11, false, false);
    __syncthreads();
  }

  // epilogue: C/D 16x16 f32 layout -> row = r + 8*half, col = l16
#pragma unroll
  for (int t = 0; t < 4; ++t) {
    int ti = t >> 1, tj = t & 1;
    f32x8 acc = (t == 0) ? acc00 : (t == 1) ? acc01 : (t == 2) ? acc10 : acc11;
    int col = n0 + wn * 32 + tj * 16 + l16;
    float bb = bias ? bias[col] : 0.0f;
#pragma unroll
    for (int r = 0; r < 8; ++r) {
      int row = m0 + wm * 32 + ti * 16 + half * 8 + r;
      if (row < M)
        C[(size_t)row * N + col] = apply_act(acc[r] + bb, act);
    }
  }
}

// ---------------- helpers ----------------
__global__ void zero_k(float* p, long n) {
  long i = (long)blockIdx.x * blockDim.x + threadIdx.x;
  if (i < n) p[i] = 0.0f;
}
__global__ void add2_k(const float* __restrict__ a, const float* __restrict__ b,
                       float* __restrict__ o, long n) {
  long i = (long)blockIdx.x * blockDim.x + threadIdx.x;
  if (i < n) o[i] = a[i] + b[i];
}

struct EmbPtrs { const float* p[9]; };

// out[row, 0:256] = sum_f embs[f][ids[row,f]]
__global__ __launch_bounds__(256) void embsum_k(EmbPtrs ep, const int* __restrict__ ids,
                                                int F, float* __restrict__ out) {
  int r = blockIdx.x, d = threadIdx.x;
  float acc = 0.f;
  for (int f = 0; f < F; ++f)
    acc += ep.p[f][(size_t)ids[r * F + f] * DG + d];
  out[(size_t)r * DG + d] = acc;
}

// LayerNorm over `dim`, optional fused gelu on input and residual add
__global__ __launch_bounds__(256) void layernorm_k(
    const float* __restrict__ in, const float* __restrict__ resid,
    const float* __restrict__ gm, const float* __restrict__ bt,
    float* __restrict__ out, int dim, float eps, int gelu_in)
{
  int r = blockIdx.x;
  __shared__ float s1[256], s2[256];
  const float* ip = in + (size_t)r * dim;
  const float* rp = resid ? resid + (size_t)r * dim : nullptr;
  float sum = 0.f, sq = 0.f;
  for (int i = threadIdx.x; i < dim; i += 256) {
    float v = ip[i];
    if (gelu_in) v = gelu_f(v);
    if (rp) v += rp[i];
    sum += v; sq += v * v;
  }
  s1[threadIdx.x] = sum; s2[threadIdx.x] = sq; __syncthreads();
  for (int st = 128; st > 0; st >>= 1) {
    if (threadIdx.x < st) { s1[threadIdx.x] += s1[threadIdx.x + st]; s2[threadIdx.x] += s2[threadIdx.x + st]; }
    __syncthreads();
  }
  float mean = s1[0] / dim;
  float var  = s2[0] / dim - mean * mean;
  float rstd = rsqrtf(var + eps);
  for (int i = threadIdx.x; i < dim; i += 256) {
    float v = ip[i];
    if (gelu_in) v = gelu_f(v);
    if (rp) v += rp[i];
    out[(size_t)r * dim + i] = (v - mean) * rstd * gm[i] + bt[i];
  }
}

// GINE message + scatter: agg[dst] += relu(h[src] + e_edge)
__global__ __launch_bounds__(256) void msg_agg_k(
    const float* __restrict__ h, const float* __restrict__ e,
    const int* __restrict__ edge_index, float* __restrict__ agg)
{
  int eid = blockIdx.x, d = threadIdx.x;
  int s  = edge_index[eid];
  int dd = edge_index[NEDGE + eid];
  float v = h[(size_t)s * DG + d] + e[(size_t)eid * DG + d];
  v = fmaxf(v, 0.0f);
  atomicAdd(&agg[(size_t)dd * DG + d], v);
}

// out[blockIdx] = dot(X[blockIdx*rowStride : +dim], w) + bias[0]
__global__ __launch_bounds__(256) void dot_k(
    const float* __restrict__ X, const float* __restrict__ w,
    const float* __restrict__ bs, float* __restrict__ out, int dim, long rowStride)
{
  __shared__ float red[256];
  const float* xp = X + (size_t)blockIdx.x * rowStride;
  float p = 0.f;
  for (int i = threadIdx.x; i < dim; i += 256) p += xp[i] * w[i];
  red[threadIdx.x] = p; __syncthreads();
  for (int st = 128; st > 0; st >>= 1) {
    if (threadIdx.x < st) red[threadIdx.x] += red[threadIdx.x + st];
    __syncthreads();
  }
  if (threadIdx.x == 0) out[blockIdx.x] = red[0] + bs[0];
}

// per-graph top-64 (iterative argmax, tie -> smallest index, like lax.top_k)
__global__ __launch_bounds__(256) void topk_k(
    const float* __restrict__ scores, const int* __restrict__ batch_vec,
    int* __restrict__ topi, float* __restrict__ gmask)
{
  int b = blockIdx.x, tid = threadIdx.x;
  __shared__ float sv[NATOM];
  __shared__ float rv[256];
  __shared__ int   ri[256];
  for (int i = tid; i < NATOM; i += 256)
    sv[i] = (batch_vec[i] == b) ? scores[i] : -1e9f;
  __syncthreads();
  for (int kk = 0; kk < TOPK; ++kk) {
    float bv = -3.4e38f; int bi = 0;
    for (int i = tid; i < NATOM; i += 256) {
      float v = sv[i];
      if (v > bv) { bv = v; bi = i; }
    }
    rv[tid] = bv; ri[tid] = bi; __syncthreads();
    for (int st = 128; st > 0; st >>= 1) {
      if (tid < st) {
        if (rv[tid + st] > rv[tid] || (rv[tid + st] == rv[tid] && ri[tid + st] < ri[tid])) {
          rv[tid] = rv[tid + st]; ri[tid] = ri[tid + st];
        }
      }
      __syncthreads();
    }
    if (tid == 0) {
      topi[b * TOPK + kk]  = ri[0];
      gmask[b * TOPK + kk] = (rv[0] > -5e8f) ? 1.0f : 0.0f;
      sv[ri[0]] = -3.4e38f;
    }
    __syncthreads();
  }
}

__global__ __launch_bounds__(256) void gtok_k(
    const float* __restrict__ h, const int* __restrict__ topi,
    const float* __restrict__ gmask, float* __restrict__ gt)
{
  int row = blockIdx.x;                 // b*64 + kk
  int i = topi[row]; float m = gmask[row];
  gt[(size_t)row * DG + threadIdx.x] = h[(size_t)i * DG + threadIdx.x] * m;
}

// attn mask vector: [1, gmask(64), 1, text_attn(128), 1]
__global__ void attnmask_k(const float* __restrict__ gmask, const int* __restrict__ tattn,
                           float* __restrict__ attn_f)
{
  int b = blockIdx.x, s = threadIdx.x;
  if (s >= SEQ) return;
  float v;
  if (s == 0 || s == 65 || s == 194) v = 1.0f;
  else if (s <= 64)  v = gmask[b * TOPK + s - 1];
  else               v = (float)tattn[b * TTXT + s - 66];
  attn_f[b * SEQ + s] = v;
}

// assemble BERT input embedding (pre-LN)
__global__ __launch_bounds__(256) void build_emb_k(
    const float* __restrict__ g, const float* __restrict__ we,
    const float* __restrict__ pe, const float* __restrict__ te,
    const int* __restrict__ ids, float* __restrict__ x)
{
  int bs = blockIdx.x;
  int b = bs / SEQ, s = bs % SEQ;
  const float* src;
  if (s == 0)        src = we + (size_t)101 * HDIM;
  else if (s <= 64)  src = g  + (size_t)(b * TOPK + (s - 1)) * HDIM;
  else if (s == 65)  src = we + (size_t)102 * HDIM;
  else if (s <= 193) src = we + (size_t)ids[b * TTXT + (s - 66)] * HDIM;
  else               src = we + (size_t)102 * HDIM;
  int ttype = (s < 66) ? 0 : 1;
  for (int i = threadIdx.x; i < HDIM; i += 256)
    x[(size_t)bs * HDIM + i] = src[i] + pe[(size_t)s * HDIM + i] + te[(size_t)ttype * HDIM + i];
}

// fused attention for one (b, head): K,V resident in LDS as bf16 pairs
__global__ __launch_bounds__(256) void attn_k(
    const float* __restrict__ q, const float* __restrict__ k, const float* __restrict__ v,
    const float* __restrict__ mask, float* __restrict__ ctx)
{
  const int bh = blockIdx.x;
  const int b = bh / NHEAD, hd = bh % NHEAD;
  const int h0 = hd * 64;
  const int tid = threadIdx.x, lane = tid & 31, wv = tid >> 5;

  __shared__ unsigned int Ks[SEQ * 32];
  __shared__ unsigned int Vs[SEQ * 32];
  __shared__ float sc[8][SEQ + 1];

  for (int i = tid; i < SEQ * 32; i += 256) {
    int s = i >> 5, dp = i & 31;
    size_t base = ((size_t)(b * SEQ + s)) * HDIM + h0 + dp * 2;
    Ks[i] = pack2bf(k[base], k[base + 1]);
    Vs[i] = pack2bf(v[base], v[base + 1]);
  }
  __syncthreads();

  const float scale = 0.125f;  // 1/sqrt(64)
  for (int qr = wv; qr < SEQ; qr += 8) {
    size_t qbase = ((size_t)(b * SEQ + qr)) * HDIM + h0;
    float q0 = q[qbase + lane];
    float q1 = q[qbase + 32 + lane];

    // pass 1: scores for this lane's k positions
    float mx = -3.4e38f;
    for (int kk = lane; kk < SEQ; kk += 32) {
      float acc = 0.f;
#pragma unroll
      for (int dp = 0; dp < 32; ++dp) {
        unsigned int kp = Ks[kk * 32 + dp];
        float k0f = bf2f(kp & 0xFFFFu);
        float k1f = bf2f(kp >> 16);
        float qa, qb;
        if (dp < 16) { qa = __shfl(q0, 2 * dp, 32);      qb = __shfl(q0, 2 * dp + 1, 32); }
        else         { qa = __shfl(q1, 2 * dp - 32, 32); qb = __shfl(q1, 2 * dp - 31, 32); }
        acc += qa * k0f + qb * k1f;
      }
      float sval = acc * scale + (1.0f - mask[b * SEQ + kk]) * (-1e4f);
      sc[wv][kk] = sval;
      mx = fmaxf(mx, sval);
    }
#pragma unroll
    for (int off = 16; off > 0; off >>= 1) mx = fmaxf(mx, __shfl_xor(mx, off, 32));

    float sum = 0.f;
    for (int kk = lane; kk < SEQ; kk += 32) {
      float e = __expf(sc[wv][kk] - mx);
      sum += e;
      sc[wv][kk] = e;
    }
#pragma unroll
    for (int off = 16; off > 0; off >>= 1) sum += __shfl_xor(sum, off, 32);
    float rinv = 1.0f / sum;

    // pass 2: ctx[d] for d = 2*lane, 2*lane+1
    float a0 = 0.f, a1 = 0.f;
    for (int kk = 0; kk < SEQ; ++kk) {
      float w = sc[wv][kk];
      unsigned int vp = Vs[kk * 32 + lane];
      a0 += w * bf2f(vp & 0xFFFFu);
      a1 += w * bf2f(vp >> 16);
    }
    ctx[qbase + 2 * lane]     = a0 * rinv;
    ctx[qbase + 2 * lane + 1] = a1 * rinv;
  }
}

// ---------------- host orchestration ----------------
extern "C" void kernel_launch(void* const* d_in, const int* in_sizes, int n_in,
                              void* d_out, int out_size, void* d_ws, size_t ws_size,
                              hipStream_t stream) {
  (void)in_sizes; (void)n_in; (void)out_size; (void)ws_size;
  auto F = [&](int i) { return (const float*)d_in[i]; };
  auto I = [&](int i) { return (const int*)d_in[i]; };

  // --- input index map (setup_inputs dict order, params depth-first insertion order) ---
  const int* x_atom    = I(0);
  const int* edge_attr = I(1);
  const int* edge_idx  = I(2);
  const int* batch_vec = I(3);
  const int* text_ids  = I(4);
  const int* text_attn = I(5);
  // params: atom_embs[9]=6..14, atom_ln=15,16, atom_mlp=17..20,
  //         bond_embs[3]=21..23, bond_ln=24,25, bond_mlp=26..29,
  //         gine[4]x(W1,b1,W2,b2,ln_g,ln_b)=30..53, score=54,55, proj=56,57,
  //         word=58, pos=59, type=60, emb_ln=61,62, bert[12]x16=63..254, cls=255,256
  const int GINE0 = 30, BERT0 = 63;

  // --- workspace bump allocator ---
  char* wsp = (char*)d_ws;
  size_t off = 0;
  auto alloc = [&](size_t bytes) -> float* {
    float* p = (float*)(wsp + off);
    off = (off + bytes + 255) & ~(size_t)255;
    return p;
  };
  float* h    = alloc((size_t)NATOM * DG * 4);
  float* h2   = alloc((size_t)NATOM * DG * 4);
  float* agg  = alloc((size_t)NATOM * DG * 4);
  float* t1   = alloc((size_t)NEDGE * DG * 4);   // shared temps (>= max rows)
  float* t2   = alloc((size_t)NEDGE * DG * 4);
  float* e    = alloc((size_t)NEDGE * DG * 4);
  float* scores = alloc((size_t)NATOM * 4);
  int*   topi   = (int*)alloc((size_t)BATCH * TOPK * 4);
  float* gmask  = alloc((size_t)BATCH * TOPK * 4);
  float* gtok   = alloc((size_t)BATCH * TOPK * DG * 4);
  float* g      = alloc((size_t)BATCH * TOPK * HDIM * 4);
  float* attnf  = alloc((size_t)BATCH * SEQ * 4);
  float* xa   = alloc((size_t)NTOK * HDIM * 4);
  float* xb   = alloc((size_t)NTOK * HDIM * 4);
  float* qb   = alloc((size_t)NTOK * HDIM * 4);
  float* kb   = alloc((size_t)NTOK * HDIM * 4);
  float* vb   = alloc((size_t)NTOK * HDIM * 4);
  float* ctxb = alloc((size_t)NTOK * HDIM * 4);
  float* tmpb = alloc((size_t)NTOK * HDIM * 4);
  float* ff   = alloc((size_t)NTOK * DFF * 4);

  // --- pre-pack all GEMM weights to bf16-pair layout (uint per K-pair) ---
  auto packw = [&](const float* W, int K, int N) -> unsigned int* {
    unsigned int* Wp = (unsigned int*)alloc((size_t)(K / 2) * N * 4);
    long total = (long)(K / 2) * N;
    packw_k<<<(int)((total + 255) / 256), 256, 0, stream>>>(W, Wp, K, N);
    return Wp;
  };
  unsigned int* atomW1 = packw(F(17), DG, DG);
  unsigned int* atomW2 = packw(F(19), DG, DG);
  unsigned int* bondW1 = packw(F(26), DG, DG);
  unsigned int* bondW2 = packw(F(28), DG, DG);
  unsigned int* gineW1[4]; unsigned int* gineW2[4];
  for (int l = 0; l < 4; ++l) {
    gineW1[l] = packw(F(GINE0 + l * 6 + 0), DG, DG);
    gineW2[l] = packw(F(GINE0 + l * 6 + 2), DG, DG);
  }
  unsigned int* projW = packw(F(56), DG, HDIM);
  unsigned int* bw[NLAYER][6];
  for (int l = 0; l < NLAYER; ++l) {
    int p0 = BERT0 + l * 16;
    bw[l][0] = packw(F(p0 + 0),  HDIM, HDIM);   // Wq
    bw[l][1] = packw(F(p0 + 2),  HDIM, HDIM);   // Wk
    bw[l][2] = packw(F(p0 + 4),  HDIM, HDIM);   // Wv
    bw[l][3] = packw(F(p0 + 6),  HDIM, HDIM);   // Wo
    bw[l][4] = packw(F(p0 + 10), HDIM, DFF);    // Wi
    bw[l][5] = packw(F(p0 + 12), DFF,  HDIM);   // Wf
  }

  auto gemm = [&](const float* A, const unsigned int* Wp, const float* bias, float* C,
                  int M, int N, int K, int act) {
    dim3 grid(N / BN, (M + BM - 1) / BM);
    gemm_wmma_k<<<grid, 256, 0, stream>>>(A, Wp, bias, C, M, N, K, act);
  };

  // ---- atom encoder: sum embs -> LN(1e-5) -> gelu MLP ----
  EmbPtrs ap; for (int i = 0; i < 9; ++i) ap.p[i] = F(6 + i);
  embsum_k<<<NATOM, 256, 0, stream>>>(ap, x_atom, 9, t1);
  layernorm_k<<<NATOM, 256, 0, stream>>>(t1, nullptr, F(15), F(16), t2, DG, 1e-5f, 0);
  gemm(t2, atomW1, F(18), t1, NATOM, DG, DG, 1);
  gemm(t1, atomW2, F(20), h,  NATOM, DG, DG, 0);

  // ---- bond encoder ----
  EmbPtrs bp; for (int i = 0; i < 3; ++i) bp.p[i] = F(21 + i);
  for (int i = 3; i < 9; ++i) bp.p[i] = nullptr;
  embsum_k<<<NEDGE, 256, 0, stream>>>(bp, edge_attr, 3, t1);
  layernorm_k<<<NEDGE, 256, 0, stream>>>(t1, nullptr, F(24), F(25), t2, DG, 1e-5f, 0);
  gemm(t2, bondW1, F(27), t1, NEDGE, DG, DG, 1);
  gemm(t1, bondW2, F(29), e,  NEDGE, DG, DG, 0);

  // ---- 4 GINE layers ----
  for (int l = 0; l < 4; ++l) {
    int p0 = GINE0 + l * 6;
    long nEl = (long)NATOM * DG;
    zero_k<<<(int)((nEl + 255) / 256), 256, 0, stream>>>(agg, nEl);
    msg_agg_k<<<NEDGE, 256, 0, stream>>>(h, e, edge_idx, agg);
    add2_k<<<(int)((nEl + 255) / 256), 256, 0, stream>>>(h, agg, t1, nEl);
    gemm(t1, gineW1[l], F(p0 + 1), t2, NATOM, DG, DG, 2);          // relu
    gemm(t2, gineW2[l], F(p0 + 3), t1, NATOM, DG, DG, 0);
    layernorm_k<<<NATOM, 256, 0, stream>>>(t1, h, F(p0 + 4), F(p0 + 5), h2, DG, 1e-5f, 1); // LN(gelu(z)+h)
    float* sw = h; h = h2; h2 = sw;
  }

  // ---- scores, top-k, graph tokens, projection ----
  dot_k<<<NATOM, 256, 0, stream>>>(h, F(54), F(55), scores, DG, DG);
  topk_k<<<BATCH, 256, 0, stream>>>(scores, batch_vec, topi, gmask);
  gtok_k<<<BATCH * TOPK, 256, 0, stream>>>(h, topi, gmask, gtok);
  gemm(gtok, projW, F(57), g, BATCH * TOPK, HDIM, DG, 0);
  attnmask_k<<<BATCH, 256, 0, stream>>>(gmask, text_attn, attnf);

  // ---- BERT embeddings ----
  build_emb_k<<<NTOK, 256, 0, stream>>>(g, F(58), F(59), F(60), text_ids, xa);
  layernorm_k<<<NTOK, 256, 0, stream>>>(xa, nullptr, F(61), F(62), xb, HDIM, 1e-12f, 0);
  float* xcur = xb; float* xalt = xa;

  // ---- 12 BERT layers ----
  for (int l = 0; l < NLAYER; ++l) {
    int p0 = BERT0 + l * 16;
    gemm(xcur, bw[l][0], F(p0 + 1), qb, NTOK, HDIM, HDIM, 0);
    gemm(xcur, bw[l][1], F(p0 + 3), kb, NTOK, HDIM, HDIM, 0);
    gemm(xcur, bw[l][2], F(p0 + 5), vb, NTOK, HDIM, HDIM, 0);
    attn_k<<<BATCH * NHEAD, 256, 0, stream>>>(qb, kb, vb, attnf, ctxb);
    gemm(ctxb, bw[l][3], F(p0 + 7), tmpb, NTOK, HDIM, HDIM, 0);
    layernorm_k<<<NTOK, 256, 0, stream>>>(tmpb, xcur, F(p0 + 8), F(p0 + 9), xalt, HDIM, 1e-12f, 0);
    { float* sw = xcur; xcur = xalt; xalt = sw; }
    gemm(xcur, bw[l][4], F(p0 + 11), ff, NTOK, DFF, HDIM, 1);      // gelu
    gemm(ff,   bw[l][5], F(p0 + 13), tmpb, NTOK, HDIM, DFF, 0);
    layernorm_k<<<NTOK, 256, 0, stream>>>(tmpb, xcur, F(p0 + 14), F(p0 + 15), xalt, HDIM, 1e-12f, 0);
    { float* sw = xcur; xcur = xalt; xalt = sw; }
  }

  // ---- classifier head on CLS token (row b*SEQ) ----
  dot_k<<<BATCH, 256, 0, stream>>>(xcur, F(255), F(256), (float*)d_out, HDIM, (long)SEQ * HDIM);
}
